// Baseline_Jinkyu_78237124264118
// MI455X (gfx1250) — matrix operand order repaired and checked by reading, not compiled, not run on hardware
//
#include <hip/hip_runtime.h>
#include <math.h>

// ---------------------------------------------------------------------------
// Shapes
// ---------------------------------------------------------------------------
#define BB 4
#define TT 100
#define DD 256
#define HH 16
#define WW 16
#define LL 256
#define HID 256
#define BT (BB*TT)           // 400 frames
#define FR 65536             // 256*256 elements per frame
#define EPS 1e-5f

typedef _Float16 v16h __attribute__((ext_vector_type(16)));
typedef _Float16 v8h  __attribute__((ext_vector_type(8)));
typedef float    v8f  __attribute__((ext_vector_type(8)));

#define WMMA16(a, b, c) \
  __builtin_amdgcn_wmma_f32_16x16x32_f16(false, (a), false, (b), (short)0, (c), false, false)

// Build a 16-element f16 fragment from two contiguous 8-element (16B) chunks.
__device__ __forceinline__ v16h load_frag(const _Float16* p0, const _Float16* p1) {
  v8h lo = *(const v8h*)p0;
  v8h hi = *(const v8h*)p1;
  return __builtin_shufflevector(lo, hi, 0,1,2,3,4,5,6,7,8,9,10,11,12,13,14,15);
}

// ---------------------------------------------------------------------------
// K0: repack conv weights to f16 [tap][co][ci]; lstm_wih to f16 [u][d]
// ---------------------------------------------------------------------------
__global__ void prep_w(const float* __restrict__ conv_w, const float* __restrict__ wih,
                       _Float16* __restrict__ wc16, _Float16* __restrict__ wih16) {
  int stride = gridDim.x * blockDim.x;
  for (int i = blockIdx.x * blockDim.x + threadIdx.x; i < 9*256*256; i += stride) {
    int ci = i & 255, co = (i >> 8) & 255, tap = i >> 16;     // i = tap*65536+co*256+ci
    int kh = tap / 3, kw = tap - kh * 3;
    wc16[i] = (_Float16)conv_w[((co*256 + ci)*3 + kh)*3 + kw];
  }
  for (int i = blockIdx.x * blockDim.x + threadIdx.x; i < 1024*256; i += stride)
    wih16[i] = (_Float16)wih[i];
}

// ---------------------------------------------------------------------------
// K0b: Wr = whh + wih @ w_w  (f16, [u][h] row-major = B-matrix layout)
//      cgate[u] = bih[u] + bhh[u] + sum_d w_b[d]*wih[u][d]
// ---------------------------------------------------------------------------
__global__ __launch_bounds__(256) void wr_k(const float* __restrict__ whh,
    const float* __restrict__ wih, const float* __restrict__ w_w,
    const float* __restrict__ w_b, const float* __restrict__ bih,
    const float* __restrict__ bhh, _Float16* __restrict__ wr16,
    float* __restrict__ cgate) {
  int u = blockIdx.x, h = threadIdx.x;
  __shared__ float row[256];
  __shared__ float part[8];
  row[h] = wih[(size_t)u*256 + h];
  __syncthreads();
  float acc = whh[(size_t)u*256 + h];
  for (int d = 0; d < 256; ++d) acc += row[d] * w_w[(size_t)d*256 + h];
  wr16[(size_t)u*256 + h] = (_Float16)acc;
  float p = row[h] * w_b[h];
  for (int o = 16; o > 0; o >>= 1) p += __shfl_down(p, o);
  if ((h & 31) == 0) part[h >> 5] = p;
  __syncthreads();
  if (h == 0) {
    float s = bih[u] + bhh[u];
    for (int i = 0; i < 8; ++i) s += part[i];
    cgate[u] = s;
  }
}

// ---------------------------------------------------------------------------
// K1: conv 3x3 as 9 shifted WMMA GEMMs per frame.
//     Block = (frame, co-slab of 64). LDS holds the frame as X[p][ci] f16
//     (+1 zero row for halo). A = weight tile [co][ci], B = X^T tile.
// ---------------------------------------------------------------------------
#define XRS 264   // f16 row stride in LDS (16B aligned, bank-spread)
__global__ __launch_bounds__(256) void conv_wmma(const float* __restrict__ x,
    const _Float16* __restrict__ w16, const float* __restrict__ bias,
    float* __restrict__ y) {
  extern __shared__ _Float16 xs[];                // 257 * XRS f16
  int f = blockIdx.x >> 2, slab = blockIdx.x & 3;
  int tid = threadIdx.x;
  const float* xf = x + (size_t)f * FR;
  // stage: xs[p][ci] = f16(x[f][ci][p])  (coalesced global reads, p = lane)
  #pragma unroll 4
  for (int ci = 0; ci < 256; ++ci)
    xs[tid * XRS + ci] = (_Float16)xf[ci*256 + tid];
  for (int i = tid; i < XRS; i += 256) xs[256 * XRS + i] = (_Float16)0.f;   // zero halo row
  __syncthreads();

  int wave = tid >> 5, lane = tid & 31;
  int mt  = wave >> 1;                 // 0..3 co-tile within slab
  int ntb = (wave & 1) * 8;            // first of 8 p-tiles
  int co0 = slab * 64 + mt * 16;
  int m = lane & 15, half = lane >> 4, n = lane & 15;

  v8f acc[8];
  #pragma unroll
  for (int j = 0; j < 8; ++j) acc[j] = {};

  for (int tap = 0; tap < 9; ++tap) {
    int dh = tap / 3 - 1, dw = tap % 3 - 1;
    int wp = n + dw;
    bool cval = (unsigned)wp < 16u;
    const _Float16* wbase = w16 + ((size_t)(tap*256 + co0 + m))*256 + half*8;
    for (int k0 = 0; k0 < 256; k0 += 32) {
      v16h afrag = load_frag(wbase + k0, wbase + k0 + 16);
      #pragma unroll
      for (int j = 0; j < 8; ++j) {
        int nt = ntb + j;                          // p-tile == image row nt
        int hp = nt + dh;
        bool rval = (unsigned)hp < 16u;
        int prow = (rval && cval) ? (hp*16 + wp) : 256;   // 256 = zero row
        const _Float16* bp = xs + prow * XRS + k0 + half*8;
        v16h bfrag = load_frag(bp, bp + 16);
        acc[j] = WMMA16(afrag, bfrag, acc[j]);
      }
    }
  }
  float* yf = y + (size_t)f * FR;
  #pragma unroll
  for (int j = 0; j < 8; ++j) {
    int p = (ntb + j) * 16 + n;
    #pragma unroll
    for (int r = 0; r < 8; ++r) {
      int co = co0 + r + half*8;
      yf[co*256 + p] = acc[j][r] + bias[co];
    }
  }
}

// ---------------------------------------------------------------------------
// K2: per-channel batch stats over 400x256 values -> mean, rstd
// ---------------------------------------------------------------------------
__global__ __launch_bounds__(256) void bn_stats(const float* __restrict__ y,
    float* __restrict__ mean, float* __restrict__ rstd) {
  int c = blockIdx.x, tid = threadIdx.x;
  float s = 0.f, s2 = 0.f;
  for (int f = 0; f < BT; ++f) {
    float v = y[(size_t)f*FR + c*256 + tid];
    s += v; s2 += v*v;
  }
  __shared__ float ls[16];
  for (int o = 16; o > 0; o >>= 1) { s += __shfl_down(s, o); s2 += __shfl_down(s2, o); }
  if ((tid & 31) == 0) { ls[tid >> 5] = s; ls[8 + (tid >> 5)] = s2; }
  __syncthreads();
  if (tid == 0) {
    float a = 0.f, b = 0.f;
    for (int i = 0; i < 8; ++i) { a += ls[i]; b += ls[8 + i]; }
    float mu = a / (float)(BT * 256);
    float var = b / (float)(BT * 256) - mu * mu;
    mean[c] = mu;
    rstd[c] = rsqrtf(var + EPS);
  }
}

// ---------------------------------------------------------------------------
// K3: apply BN+ReLU in place; plog[bt][c] = sum_d feats * wattn[d]
// ---------------------------------------------------------------------------
__global__ __launch_bounds__(256) void bn_apply(const float* __restrict__ gamma,
    const float* __restrict__ beta, const float* __restrict__ mean,
    const float* __restrict__ rstd, const float* __restrict__ wattn,
    float* __restrict__ y, float* __restrict__ plog) {
  int bt = blockIdx.x, d = threadIdx.x;
  __shared__ float part[8];
  float wa = wattn[d];
  float* yb = y + (size_t)bt * FR;
  for (int c = 0; c < 256; ++c) {
    float v = yb[c*256 + d];
    v = gamma[c] * (v - mean[c]) * rstd[c] + beta[c];
    v = fmaxf(v, 0.f);
    yb[c*256 + d] = v;
    float p = v * wa;
    for (int o = 16; o > 0; o >>= 1) p += __shfl_down(p, o);
    if ((d & 31) == 0) part[d >> 5] = p;
    __syncthreads();
    if (d == 0) {
      float s = 0.f;
      for (int i = 0; i < 8; ++i) s += part[i];
      plog[(size_t)bt*256 + c] = s;
    }
    __syncthreads();
  }
}

// ---------------------------------------------------------------------------
// K4: softmax over l. hx-dependent logit term is constant over l -> cancels.
//     Writes alphas & alpha_logps straight into d_out, plus ws copy of alpha.
// ---------------------------------------------------------------------------
__global__ __launch_bounds__(256) void softmax_k(const float* __restrict__ plog,
    float* __restrict__ out, float* __restrict__ alpha_ws) {
  int bt = blockIdx.x, b = bt / TT, t = bt % TT, l = threadIdx.x;
  __shared__ float sm[8];
  float x = plog[(size_t)bt*256 + l];
  float m = x;
  for (int o = 16; o > 0; o >>= 1) m = fmaxf(m, __shfl_xor(m, o));
  if ((l & 31) == 0) sm[l >> 5] = m;
  __syncthreads();
  m = sm[0];
  for (int i = 1; i < 8; ++i) m = fmaxf(m, sm[i]);
  __syncthreads();
  float e = __expf(x - m);
  float s = e;
  for (int o = 16; o > 0; o >>= 1) s += __shfl_xor(s, o);
  if ((l & 31) == 0) sm[l >> 5] = s;
  __syncthreads();
  s = 0.f;
  for (int i = 0; i < 8; ++i) s += sm[i];
  float a = e / s;
  float lp = (x - m) - logf(s);
  size_t row = (size_t)(t*4 + b) * 256 + l;
  out[800 + row] = a;                 // alphas   [T,B,L]
  out[800 + 102400 + row] = lp;       // logalphas[T,B,L]
  alpha_ws[(size_t)bt*256 + l] = a;
}

// ---------------------------------------------------------------------------
// K5: ctxf[r=(t*4+b)][d] = sum_c alpha[bt][c] * feats[bt][c][d]  (f16 A-matrix)
// ---------------------------------------------------------------------------
__global__ __launch_bounds__(256) void ctxf_k(const float* __restrict__ y,
    const float* __restrict__ alpha, _Float16* __restrict__ ctxf16) {
  int bt = blockIdx.x, b = bt / TT, t = bt % TT, d = threadIdx.x;
  __shared__ float al[256];
  al[d] = alpha[(size_t)bt*256 + d];
  __syncthreads();
  const float* yb = y + (size_t)bt * FR;
  float acc = 0.f;
  for (int c = 0; c < 256; ++c) acc += al[c] * yb[c*256 + d];
  ctxf16[((size_t)(t*4 + b))*256 + d] = (_Float16)acc;
}

// ---------------------------------------------------------------------------
// K6: pre[r][u] = (ctxf @ wih^T)[r][u] + cgate[u]   (400x1024x256 WMMA GEMM)
// ---------------------------------------------------------------------------
__global__ __launch_bounds__(256) void pre_wmma(const _Float16* __restrict__ ctxf16,
    const _Float16* __restrict__ wih16, const float* __restrict__ cgate,
    float* __restrict__ pre) {
  int mt = blockIdx.x / 8, ng = blockIdx.x % 8;
  int wave = threadIdx.x >> 5, lane = threadIdx.x & 31;
  int nt = ng * 8 + wave;
  int m = lane & 15, half = lane >> 4, n = lane & 15;
  v8f acc = {};
  const _Float16* ab = ctxf16 + ((size_t)(mt*16 + m))*256 + half*8;
  const _Float16* bb = wih16 + ((size_t)(nt*16 + n))*256 + half*8;
  for (int k0 = 0; k0 < 256; k0 += 32) {
    v16h a = load_frag(ab + k0, ab + k0 + 16);
    v16h bf = load_frag(bb + k0, bb + k0 + 16);
    acc = WMMA16(a, bf, acc);
  }
  int u = nt*16 + n;
  float cg = cgate[u];
  #pragma unroll
  for (int r = 0; r < 8; ++r) {
    int row = mt*16 + r + half*8;
    pre[(size_t)row*1024 + u] = acc[r] + cg;
  }
}

// ---------------------------------------------------------------------------
// K7: the serial recurrence. gates = pre[t] + hx @ Wr^T via WMMA (M=16 pads B=4),
//     hx kept as f16 A-matrix in LDS, cx in registers (thread tid owns hid=tid).
// ---------------------------------------------------------------------------
#define HRS 264
__global__ __launch_bounds__(256) void rnn_k(const float* __restrict__ y,
    const float* __restrict__ pre, const _Float16* __restrict__ wr16,
    const float* __restrict__ ihw, const float* __restrict__ ihb,
    const float* __restrict__ icw, const float* __restrict__ icb,
    float* __restrict__ hxout) {
  __shared__ _Float16 hx16[16 * HRS];
  __shared__ float x0s[4 * 256];
  __shared__ float gl[4 * 1024];
  int tid = threadIdx.x;
  for (int i = tid; i < 16 * HRS; i += 256) hx16[i] = (_Float16)0.f;
  // x0[b][d] = mean over channel of feats frame (b, t=0)
  for (int b = 0; b < 4; ++b) {
    const float* yb = y + (size_t)(b * TT) * FR;
    float s = 0.f;
    for (int c = 0; c < 256; ++c) s += yb[c*256 + tid];
    x0s[b*256 + tid] = s * (1.f / 256.f);
  }
  __syncthreads();
  float cx[4];
  {
    const float* wh = ihw + (size_t)tid * 256;
    const float* wc = icw + (size_t)tid * 256;
    float hb = ihb[tid], cb = icb[tid];
    for (int b = 0; b < 4; ++b) {
      float sh = hb, sc = cb;
      for (int d = 0; d < 256; ++d) {
        float xv = x0s[b*256 + d];
        sh += xv * wh[d]; sc += xv * wc[d];
      }
      cx[b] = tanhf(sc);
      hx16[b*HRS + tid] = (_Float16)tanhf(sh);
    }
  }
  __syncthreads();

  int wave = tid >> 5, lane = tid & 31, m = lane & 15, half = lane >> 4, n = lane & 15;
  for (int t = 0; t < TT; ++t) {
    const float* pt = pre + (size_t)t * 4096;
    #pragma unroll
    for (int j = 0; j < 8; ++j) {
      int nt = wave * 8 + j;
      int u = nt * 16 + n;
      v8f acc;
      #pragma unroll
      for (int r = 0; r < 8; ++r) {
        int row = r + half * 8;
        acc[r] = (row < 4) ? pt[row*1024 + u] : 0.f;
      }
      const _Float16* bb = wr16 + (size_t)u * 256 + half*8;
      const _Float16* ab = hx16 + m * HRS + half*8;
      #pragma unroll
      for (int k0 = 0; k0 < 256; k0 += 32) {
        v16h a = load_frag(ab + k0, ab + k0 + 16);
        v16h bf = load_frag(bb + k0, bb + k0 + 16);
        acc = WMMA16(a, bf, acc);
      }
      if (half == 0) {
        #pragma unroll
        for (int r = 0; r < 4; ++r) gl[r*1024 + u] = acc[r];
      }
    }
    __syncthreads();
    float hnew[4];
    #pragma unroll
    for (int b = 0; b < 4; ++b) {
      float ig = gl[b*1024 + tid];
      float fg = gl[b*1024 + 256 + tid];
      float gg = gl[b*1024 + 512 + tid];
      float og = gl[b*1024 + 768 + tid];
      float si = 1.f / (1.f + __expf(-ig));
      float sf = 1.f / (1.f + __expf(-fg));
      float so = 1.f / (1.f + __expf(-og));
      float c2 = sf * cx[b] + si * tanhf(gg);
      cx[b] = c2;
      float h = so * tanhf(c2);
      hnew[b] = h;
      hxout[((size_t)t*4 + b)*256 + tid] = h;
    }
    __syncthreads();
    #pragma unroll
    for (int b = 0; b < 4; ++b) hx16[b*HRS + tid] = (_Float16)hnew[b];
    __syncthreads();
  }
}

// ---------------------------------------------------------------------------
// K8: preds[t][b][o] = hx[t][b] . wout[o] + wout_b[o]
// ---------------------------------------------------------------------------
__global__ void pred_k(const float* __restrict__ hx, const float* __restrict__ woutw,
                       const float* __restrict__ woutb, float* __restrict__ out) {
  int gid = blockIdx.x * blockDim.x + threadIdx.x;
  if (gid >= 800) return;
  int o = gid & 1, b = (gid >> 1) & 3, t = gid >> 3;
  const float* h = hx + ((size_t)t*4 + b) * 256;
  const float* w = woutw + o * 256;
  float s = woutb[o];
  for (int d = 0; d < 256; ++d) s += h[d] * w[d];
  out[((size_t)t*4 + b)*2 + o] = s;
}

// ---------------------------------------------------------------------------
extern "C" void kernel_launch(void* const* d_in, const int* in_sizes, int n_in,
                              void* d_out, int out_size, void* d_ws, size_t ws_size,
                              hipStream_t stream) {
  const float* cam   = (const float*)d_in[0];
  const float* convw = (const float*)d_in[1];
  const float* convb = (const float*)d_in[2];
  const float* gamma = (const float*)d_in[3];
  const float* beta  = (const float*)d_in[4];
  const float* ihw   = (const float*)d_in[5];
  const float* ihb   = (const float*)d_in[6];
  const float* icw   = (const float*)d_in[7];
  const float* icb   = (const float*)d_in[8];
  const float* w_w   = (const float*)d_in[9];
  const float* w_b   = (const float*)d_in[10];
  const float* wattn = (const float*)d_in[11];
  // d_in[12] (wattn_b) cancels inside softmax/log_softmax -> unused
  const float* wih   = (const float*)d_in[13];
  const float* whh   = (const float*)d_in[14];
  const float* bih   = (const float*)d_in[15];
  const float* bhh   = (const float*)d_in[16];
  const float* woutw = (const float*)d_in[17];
  const float* woutb = (const float*)d_in[18];
  float* out = (float*)d_out;

  // workspace layout (all offsets 1KB-aligned); total ~110 MB
  char* ws = (char*)d_ws;
  size_t off = 0;
  float*    ws_y   = (float*)(ws + off);      off += (size_t)BT * FR * 4;   // conv out / feats
  _Float16* wc16   = (_Float16*)(ws + off);   off += 9*256*256*2;
  _Float16* wih16  = (_Float16*)(ws + off);   off += 1024*256*2;
  _Float16* wr16   = (_Float16*)(ws + off);   off += 1024*256*2;
  _Float16* ctxf16 = (_Float16*)(ws + off);   off += (size_t)BT*256*2;
  float*    ws_mu  = (float*)(ws + off);      off += 1024;
  float*    ws_rs  = (float*)(ws + off);      off += 1024;
  float*    cgate  = (float*)(ws + off);      off += 4096;
  float*    plog   = (float*)(ws + off);      off += (size_t)BT*256*4;
  float*    alpha  = (float*)(ws + off);      off += (size_t)BT*256*4;
  float*    pre    = (float*)(ws + off);      off += (size_t)BT*1024*4;
  float*    hxbuf  = (float*)(ws + off);      off += (size_t)BT*256*4;

  prep_w<<<256, 256, 0, stream>>>(convw, wih, wc16, wih16);
  wr_k<<<1024, 256, 0, stream>>>(whh, wih, w_w, w_b, bih, bhh, wr16, cgate);
  conv_wmma<<<BT*4, 256, 257*XRS*sizeof(_Float16), stream>>>(cam, wc16, convb, ws_y);
  bn_stats<<<256, 256, 0, stream>>>(ws_y, ws_mu, ws_rs);
  bn_apply<<<BT, 256, 0, stream>>>(gamma, beta, ws_mu, ws_rs, wattn, ws_y, plog);
  softmax_k<<<BT, 256, 0, stream>>>(plog, out, alpha);
  ctxf_k<<<BT, 256, 0, stream>>>(ws_y, alpha, ctxf16);
  pre_wmma<<<200, 256, 0, stream>>>(ctxf16, wih16, cgate, pre);
  rnn_k<<<1, 256, 0, stream>>>(ws_y, pre, wr16, ihw, ihb, icw, icb, hxbuf);
  pred_k<<<4, 256, 0, stream>>>(hxbuf, woutw, woutb, out);
}
